// RWKVBlock_59536836657391
// MI455X (gfx1250) — matrix operand order repaired and checked by reading, not compiled, not run on hardware
//
#include <hip/hip_runtime.h>
#include <hip/hip_bf16.h>

// ---------------------------------------------------------------------------
// RWKV-style block for MI455X (gfx1250), wave32 + WMMA f32_16x16x32_f16.
//  - GEMMs: 64(M)x128(N) block tile, 32x32 register tile per wave (4 WMMAs /
//    K-step), double-buffered LDS fed by GLOBAL_LOAD_ASYNC_TO_LDS_B128
//    (s_wait_asynccnt pipelining), fallback load+ds_store + prefetch.
//  - WKV recurrence: 3-pass blocked linear scan (524K-way parallel) with the
//    sigmoid(r) gate + f16 convert fused into pass 3.
//  - LN + token-shift + mu-mix fused into a single kernel (no xa buffer).
// ---------------------------------------------------------------------------

typedef _Float16 h8   __attribute__((ext_vector_type(8)));
typedef _Float16 v16h __attribute__((ext_vector_type(16)));
typedef float    v8f  __attribute__((ext_vector_type(8)));
typedef int      v4i  __attribute__((ext_vector_type(4)));

#define DEV_INLINE __device__ __forceinline__

#if defined(__gfx1250__) && __has_builtin(__builtin_amdgcn_global_load_async_to_lds_b128)
#define HAVE_ASYNC_LDS 1
#else
#define HAVE_ASYNC_LDS 0
#endif

// Builtin param 1 is `int32x4 addrspace(1)*` (from clang diagnostic).
#define GLOBAL_V4I(p) ((__attribute__((address_space(1))) v4i*)(p))
#define LDS_V4I(p)    ((__attribute__((address_space(3))) v4i*)(p))

DEV_INLINE float sigmoidf_(float x) { return 1.0f / (1.0f + __expf(-x)); }

#if HAVE_ASYNC_LDS
template <int N>
DEV_INLINE void wait_async() {
#if __has_builtin(__builtin_amdgcn_s_wait_asynccnt)
  __builtin_amdgcn_s_wait_asynccnt(N);
#else
  asm volatile("s_wait_asynccnt %0" ::"i"(N) : "memory");
#endif
}
#endif

// ---------------------------------------------------------------------------
// Weight fp32 -> f16
// ---------------------------------------------------------------------------
__global__ __launch_bounds__(256) void cvt_f16_kernel(
    const float* __restrict__ w, _Float16* __restrict__ w16, int n) {
  int i = blockIdx.x * 256 + threadIdx.x;
  if (i < n) w16[i] = (_Float16)w[i];
}

// ---------------------------------------------------------------------------
// Fused LayerNorm + token-shift + mu-mix.
// One block per token n: computes LN(x[n]) and LN(x[n-1]) (zero at t==0) and
// writes up to 3 mixed f16 activations. Avoids materializing LN output.
// ---------------------------------------------------------------------------
__global__ __launch_bounds__(256) void ln_mix_kernel(
    const float* __restrict__ x, const float* __restrict__ lw,
    const float* __restrict__ lbias,
    const float* __restrict__ mu0, const float* __restrict__ mu1,
    const float* __restrict__ mu2,
    _Float16* __restrict__ o0, _Float16* __restrict__ o1,
    _Float16* __restrict__ o2, int T, int D) {
  const size_t n = blockIdx.x;
  const int t = (int)(n & (size_t)(T - 1));
  const float* xc = x + n * (size_t)D;
  const float* xp = xc - D;  // valid only when t > 0

  float s = 0.f, s2 = 0.f, p = 0.f, p2 = 0.f;
  for (int i = threadIdx.x; i < D; i += 256) {
    float v = xc[i];
    s += v; s2 += v * v;
    if (t) { float u = xp[i]; p += u; p2 += u * u; }
  }
  for (int off = 16; off; off >>= 1) {
    s  += __shfl_xor(s, off);  s2 += __shfl_xor(s2, off);
    p  += __shfl_xor(p, off);  p2 += __shfl_xor(p2, off);
  }
  __shared__ float red[4][8];
  const int wid = threadIdx.x >> 5, lane = threadIdx.x & 31;
  if (lane == 0) { red[0][wid] = s; red[1][wid] = s2; red[2][wid] = p; red[3][wid] = p2; }
  __syncthreads();
  if (threadIdx.x == 0) {
    float a = 0, bb = 0, c = 0, dd = 0;
    for (int i = 0; i < 8; ++i) { a += red[0][i]; bb += red[1][i]; c += red[2][i]; dd += red[3][i]; }
    red[0][0] = a; red[1][0] = bb; red[2][0] = c; red[3][0] = dd;
  }
  __syncthreads();
  const float invD = 1.f / (float)D;
  const float mc = red[0][0] * invD;
  const float rc = rsqrtf(red[1][0] * invD - mc * mc + 1e-5f);
  float mp = 0.f, rp = 0.f;
  if (t) {
    mp = red[2][0] * invD;
    rp = rsqrtf(red[3][0] * invD - mp * mp + 1e-5f);
  }
  for (int i = threadIdx.x; i < D; i += 256) {
    const float cur  = (xc[i] - mc) * rc * lw[i] + lbias[i];
    const float prev = t ? (xp[i] - mp) * rp * lw[i] + lbias[i] : 0.f;
    const size_t o = n * (size_t)D + i;
    { float m = mu0[i]; o0[o] = (_Float16)(cur * m + prev * (1.f - m)); }
    { float m = mu1[i]; o1[o] = (_Float16)(cur * m + prev * (1.f - m)); }
    if (o2) { float m = mu2[i]; o2[o] = (_Float16)(cur * m + prev * (1.f - m)); }
  }
}

// ---------------------------------------------------------------------------
// WMMA GEMM:  out[n,e] = epi( sum_k A[n,k] * W[e,k] )
//   A: [Ntok,K] f16, W: [E,K] f16 (row-major; einsum 'nk,ek->ne').
//   Block 256 thr = 8 waves; tile 64(M) x 128(N) x 32(K); wave tile 32x32
//   (2 M-frags x 2 N-frags -> 4 WMMAs per K-step). Double-buffered LDS.
// Epilogues: 0 = f32 store, 1 = exp, 2 = +residual, 3 = relu^2 -> f16 store.
// ---------------------------------------------------------------------------
template <int EPI>
__global__ __launch_bounds__(256) void gemm_wmma_kernel(
    const _Float16* __restrict__ A, const _Float16* __restrict__ W,
    const float* __restrict__ res, float* __restrict__ outF,
    _Float16* __restrict__ outH, int Ntok, int E, int K) {
  __shared__ _Float16 lA[2][64][40];   // padded rows (banks)
  __shared__ _Float16 lB[2][128][40];

  const int n0 = blockIdx.x * 128;  // output-feature tile
  const int m0 = blockIdx.y * 64;   // token tile
  const int tid = threadIdx.x;
  const int lane = tid & 31;
  const int w = tid >> 5;
  const int wm = (w & 1) * 32;   // wave M offset in tile
  const int wn = (w >> 1) * 32;  // wave N offset in tile

  // staging: thread -> (row, 8-half chunk); A once, B twice per thread
  const int ar = tid >> 2;
  const int ac = (tid & 3) * 8;

  auto stage = [&](int k0, int buf) {
    const _Float16* gA  = A + (size_t)(m0 + ar) * K + k0 + ac;
    const _Float16* gB0 = W + (size_t)(n0 + ar) * K + k0 + ac;
    const _Float16* gB1 = W + (size_t)(n0 + 64 + ar) * K + k0 + ac;
#if HAVE_ASYNC_LDS
    __builtin_amdgcn_global_load_async_to_lds_b128(
        GLOBAL_V4I(gA), LDS_V4I(&lA[buf][ar][ac]), 0, 0);
    __builtin_amdgcn_global_load_async_to_lds_b128(
        GLOBAL_V4I(gB0), LDS_V4I(&lB[buf][ar][ac]), 0, 0);
    __builtin_amdgcn_global_load_async_to_lds_b128(
        GLOBAL_V4I(gB1), LDS_V4I(&lB[buf][64 + ar][ac]), 0, 0);
#else
    *(h8*)&lA[buf][ar][ac]      = *(const h8*)gA;
    *(h8*)&lB[buf][ar][ac]      = *(const h8*)gB0;
    *(h8*)&lB[buf][64 + ar][ac] = *(const h8*)gB1;
#endif
  };

  v8f acc[2][2] = {};

  // fragment gather indices (ISA 7.12.2 layouts)
  const int am = wm + (lane & 15);
  const int kh = (lane >> 4) * 8;   // A half-wave K split
  const int ks = (lane >> 4) * 16;  // unused; B gathers also use kh
  const int bn = wn + (lane & 15);

  const int nsteps = K >> 5;
  stage(0, 0);
  for (int s = 0; s < nsteps; ++s) {
    const int buf = s & 1;
    if (s + 1 < nsteps) {
      stage((s + 1) << 5, buf ^ 1);
#if HAVE_ASYNC_LDS
      wait_async<3>();  // tile s's 3 copies done; tile s+1 in flight
#endif
    } else {
#if HAVE_ASYNC_LDS
      wait_async<0>();
#endif
    }
#if !HAVE_ASYNC_LDS
    if (s + 2 < nsteps) {  // global_prefetch_b8 two tiles ahead
      __builtin_prefetch(A + (size_t)(m0 + ar) * K + ((s + 2) << 5) + ac, 0, 1);
      __builtin_prefetch(W + (size_t)(n0 + ar) * K + ((s + 2) << 5) + ac, 0, 1);
    }
#endif
    __syncthreads();

    union U { v16h v; h8 h[2]; };
    U aU[2], bU[2];
#pragma unroll
    for (int mf = 0; mf < 2; ++mf) {
      aU[mf].h[0] = *(const h8*)&lA[buf][am + mf * 16][kh];
      aU[mf].h[1] = *(const h8*)&lA[buf][am + mf * 16][16 + kh];
    }
#pragma unroll
    for (int nf = 0; nf < 2; ++nf) {
      bU[nf].h[0] = *(const h8*)&lB[buf][bn + nf * 16][kh];
      bU[nf].h[1] = *(const h8*)&lB[buf][bn + nf * 16][16 + kh];
    }
#pragma unroll
    for (int mf = 0; mf < 2; ++mf)
#pragma unroll
      for (int nf = 0; nf < 2; ++nf)
        acc[mf][nf] = __builtin_amdgcn_wmma_f32_16x16x32_f16(
            false, aU[mf].v, false, bU[nf].v, (short)0, acc[mf][nf], false,
            false);
    __syncthreads();
  }

  // C/D layout: VGPR j holds row (j + 8*(lane>=16)), col = lane&15
  const int colb = n0 + wn + (lane & 15);
  const int rowb = m0 + wm + ((lane >> 4) << 3);
#pragma unroll
  for (int mf = 0; mf < 2; ++mf)
#pragma unroll
    for (int nf = 0; nf < 2; ++nf)
#pragma unroll
      for (int j = 0; j < 8; ++j) {
        const size_t i0 = (size_t)(rowb + mf * 16 + j) * E + colb + nf * 16;
        const float a = acc[mf][nf][j];
        if constexpr (EPI == 0) {
          outF[i0] = a;
        } else if constexpr (EPI == 1) {
          outF[i0] = __expf(a);
        } else if constexpr (EPI == 2) {
          outF[i0] = res[i0] + a;
        } else {
          const float r_ = fmaxf(a, 0.f);
          outH[i0] = (_Float16)(r_ * r_);
        }
      }
}

// ---------------------------------------------------------------------------
// WKV linear recurrence as a 3-pass blocked scan.
//   state_t = decay*state_{t-1} + in_t  (c with kv, n with k)
// Pass1: per-(b,d,chunk) local scan from zero -> chunk summaries.
// Pass2: per-(b,d) sequential combine of 64 summaries -> incoming prefixes.
// Pass3: per-(b,d,chunk) exact re-scan; emits g = f16(sigmoid(r)*wkv) fused.
// ---------------------------------------------------------------------------
__global__ __launch_bounds__(256) void wkv_pass1_kernel(
    const float* __restrict__ Kb, const float* __restrict__ Vb,
    const float* __restrict__ log_decay, float* __restrict__ Cc,
    float* __restrict__ Cn, int T, int D, int L, int C) {
  const int gid = blockIdx.x * 256 + threadIdx.x;  // B*C*D lanes
  const int d = gid & (D - 1);
  const int bc = gid / D;
  const int ch = bc & (C - 1);
  const int b = bc / C;
  const float decay = __expf(-__expf(log_decay[d]));
  float c = 0.f, n = 0.f;
  size_t idx = ((size_t)b * T + (size_t)ch * L) * D + d;
  for (int t = 0; t < L; ++t, idx += D) {
    const float k = Kb[idx];
    const float kv = k * Vb[idx];
    c = c * decay + kv;
    n = n * decay + k;
  }
  Cc[gid] = c;
  Cn[gid] = n;
}

__global__ __launch_bounds__(256) void wkv_pass2_kernel(
    const float* __restrict__ log_decay, float* __restrict__ Cc,
    float* __restrict__ Cn, int D, int L, int C) {
  const int gid = blockIdx.x * 256 + threadIdx.x;  // B*D lanes
  const int d = gid & (D - 1);
  const int b = gid / D;
  const float dL = __expf(-(float)L * __expf(log_decay[d]));  // decay^L
  float rc = 0.f, rn = 0.f;
  size_t base = (size_t)b * C * D + d;
  for (int ch = 0; ch < C; ++ch) {
    const size_t i = base + (size_t)ch * D;
    const float lc = Cc[i], ln_ = Cn[i];
    Cc[i] = rc;  // incoming prefix for this chunk
    Cn[i] = rn;
    rc = dL * rc + lc;
    rn = dL * rn + ln_;
  }
}

__global__ __launch_bounds__(256) void wkv_pass3_kernel(
    const float* __restrict__ Kb, const float* __restrict__ Vb,
    const float* __restrict__ Rb, const float* __restrict__ log_gain,
    const float* __restrict__ log_decay, const float* __restrict__ Cc,
    const float* __restrict__ Cn, _Float16* __restrict__ g, int T, int D,
    int L, int C) {
  const int gid = blockIdx.x * 256 + threadIdx.x;  // B*C*D lanes
  const int d = gid & (D - 1);
  const int bc = gid / D;
  const int ch = bc & (C - 1);
  const int b = bc / C;
  const float decay = __expf(-__expf(log_decay[d]));
  const float gain = __expf(log_gain[d]) - 1.f;
  float c = Cc[gid], n = Cn[gid];
  size_t idx = ((size_t)b * T + (size_t)ch * L) * D + d;
  for (int t = 0; t < L; ++t, idx += D) {
    const float k = Kb[idx];
    const float v = Vb[idx];
    const float kv = k * v;
    c = c * decay + kv;
    n = n * decay + k;
    const float out = (c + gain * kv) / (n + gain * k);
    g[idx] = (_Float16)(sigmoidf_(Rb[idx]) * out);
  }
}

// out = x2 + sigmoid(rr) * mlp
__global__ __launch_bounds__(256) void final_kernel(
    const float* __restrict__ x2, const float* __restrict__ rr,
    const float* __restrict__ mlp, float* __restrict__ out) {
  size_t i = (size_t)blockIdx.x * 256 + threadIdx.x;
  out[i] = x2[i] + sigmoidf_(rr[i]) * mlp[i];
}

// ---------------------------------------------------------------------------
// Host-side orchestration
// ---------------------------------------------------------------------------
extern "C" void kernel_launch(void* const* d_in, const int* in_sizes, int n_in,
                              void* d_out, int out_size, void* d_ws,
                              size_t ws_size, hipStream_t stream) {
  constexpr int B = 8, T = 4096, D = 1024, H = 4096;
  constexpr int SC = 64;            // scan chunks per sequence
  constexpr int SL = T / SC;        // chunk length (64)
  constexpr size_t N = (size_t)B * T;
  constexpr size_t ND = N * D;
  constexpr size_t MiB = 1ull << 20;

  const float* x        = (const float*)d_in[0];
  const float* ln1_w    = (const float*)d_in[1];
  const float* ln1_b    = (const float*)d_in[2];
  const float* mu_k     = (const float*)d_in[3];
  const float* mu_v     = (const float*)d_in[4];
  const float* mu_r     = (const float*)d_in[5];
  const float* Wk       = (const float*)d_in[6];
  const float* Wv       = (const float*)d_in[7];
  const float* Wr       = (const float*)d_in[8];
  const float* Wo       = (const float*)d_in[9];
  const float* log_gain = (const float*)d_in[10];
  const float* log_decay= (const float*)d_in[11];
  const float* ln2_w    = (const float*)d_in[12];
  const float* ln2_b    = (const float*)d_in[13];
  const float* mu_m     = (const float*)d_in[14];
  const float* mu_mr    = (const float*)d_in[15];
  const float* Wmk      = (const float*)d_in[16];
  const float* Wmv      = (const float*)d_in[17];
  const float* Wmr      = (const float*)d_in[18];
  float* out = (float*)d_out;

  char* ws = (char*)d_ws;
  if (ws_size < 864 * MiB) return;  // deterministic guard

  // --- f16 weight pool + scan summaries (first 32 MiB) ---
  _Float16* Wk16  = (_Float16*)(ws + 0 * MiB);
  _Float16* Wv16  = (_Float16*)(ws + 2 * MiB);
  _Float16* Wr16  = (_Float16*)(ws + 4 * MiB);
  _Float16* Wo16  = (_Float16*)(ws + 6 * MiB);
  _Float16* Wmr16 = (_Float16*)(ws + 8 * MiB);
  _Float16* Wmk16 = (_Float16*)(ws + 10 * MiB);
  _Float16* Wmv16 = (_Float16*)(ws + 18 * MiB);
  float*    Cc    = (float*)(ws + 26 * MiB);  // [B, SC, D] = 2 MiB
  float*    Cn    = (float*)(ws + 28 * MiB);  // [B, SC, D] = 2 MiB

  char* oA = ws + 32 * MiB;    // 128 MiB: K -> x2
  char* oB = ws + 160 * MiB;   // 128 MiB: V -> hidden(lo)
  char* oC = ws + 288 * MiB;   // 128 MiB: R -> hidden(hi)
  char* oD = ws + 416 * MiB;   // 192 MiB: mk|mv|mr -> mm|mmr|g
  char* oE = ws + 608 * MiB;   // 128 MiB: mlp
  char* oF = ws + 736 * MiB;   // 128 MiB: rr

  _Float16* mk   = (_Float16*)oD;
  _Float16* mv   = (_Float16*)(oD + 64 * MiB);
  _Float16* mr   = (_Float16*)(oD + 128 * MiB);
  float*    Kbuf = (float*)oA;
  float*    Vbuf = (float*)oB;
  float*    Rbuf = (float*)oC;
  _Float16* g    = (_Float16*)(oD + 128 * MiB);  // overwrites mr (dead)
  float*    x2   = (float*)oA;
  _Float16* mm   = (_Float16*)oD;
  _Float16* mmr  = (_Float16*)(oD + 64 * MiB);
  _Float16* hid  = (_Float16*)oB;  // N*H f16 = 256 MiB spans oB+oC
  float*    mlp  = (float*)oE;
  float*    rr   = (float*)oF;

  const int EW = (int)(ND / 256);
  const dim3 gD(D / 128, (int)(N / 64));   // GEMM out N x D
  const dim3 gH(H / 128, (int)(N / 64));   // GEMM out N x H
  const int scanGrid = (B * SC * D) / 256; // 2048 blocks

  // 1) weights -> f16
  cvt_f16_kernel<<<(D * D + 255) / 256, 256, 0, stream>>>(Wk, Wk16, D * D);
  cvt_f16_kernel<<<(D * D + 255) / 256, 256, 0, stream>>>(Wv, Wv16, D * D);
  cvt_f16_kernel<<<(D * D + 255) / 256, 256, 0, stream>>>(Wr, Wr16, D * D);
  cvt_f16_kernel<<<(D * D + 255) / 256, 256, 0, stream>>>(Wo, Wo16, D * D);
  cvt_f16_kernel<<<(D * D + 255) / 256, 256, 0, stream>>>(Wmr, Wmr16, D * D);
  cvt_f16_kernel<<<(H * D + 255) / 256, 256, 0, stream>>>(Wmk, Wmk16, H * D);
  cvt_f16_kernel<<<(D * H + 255) / 256, 256, 0, stream>>>(Wmv, Wmv16, D * H);

  // 2) fused LN1 + shift + mix -> mk, mv, mr (f16)
  ln_mix_kernel<<<(int)N, 256, 0, stream>>>(x, ln1_w, ln1_b, mu_k, mu_v, mu_r,
                                            mk, mv, mr, T, D);

  // 3) K = exp(mk Wk^T), V = mv Wv^T, R = mr Wr^T
  gemm_wmma_kernel<1><<<gD, 256, 0, stream>>>(mk, Wk16, nullptr, Kbuf, nullptr,
                                              (int)N, D, D);
  gemm_wmma_kernel<0><<<gD, 256, 0, stream>>>(mv, Wv16, nullptr, Vbuf, nullptr,
                                              (int)N, D, D);
  gemm_wmma_kernel<0><<<gD, 256, 0, stream>>>(mr, Wr16, nullptr, Rbuf, nullptr,
                                              (int)N, D, D);

  // 4) blocked WKV scan; pass3 fuses sigmoid(r)*wkv -> g (f16)
  wkv_pass1_kernel<<<scanGrid, 256, 0, stream>>>(Kbuf, Vbuf, log_decay, Cc, Cn,
                                                 T, D, SL, SC);
  wkv_pass2_kernel<<<(B * D) / 256, 256, 0, stream>>>(log_decay, Cc, Cn, D, SL,
                                                      SC);
  wkv_pass3_kernel<<<scanGrid, 256, 0, stream>>>(Kbuf, Vbuf, Rbuf, log_gain,
                                                 log_decay, Cc, Cn, g, T, D,
                                                 SL, SC);

  // 5) x2 = x + g Wo^T (residual fused)
  gemm_wmma_kernel<2><<<gD, 256, 0, stream>>>(g, Wo16, x, x2, nullptr, (int)N,
                                              D, D);

  // 6) fused LN2 + shift + mix -> mm, mmr
  ln_mix_kernel<<<(int)N, 256, 0, stream>>>(x2, ln2_w, ln2_b, mu_m, mu_mr,
                                            nullptr, mm, mmr, nullptr, T, D);

  // 7) hidden = relu(mm Wmk^T)^2 -> f16 (fused epilogue)
  gemm_wmma_kernel<3><<<gH, 256, 0, stream>>>(mm, Wmk16, nullptr, nullptr, hid,
                                              (int)N, H, D);
  // 8) mlp = hidden Wmv^T ; rr = mmr Wmr^T
  gemm_wmma_kernel<0><<<gD, 256, 0, stream>>>(hid, Wmv16, nullptr, mlp, nullptr,
                                              (int)N, D, H);
  gemm_wmma_kernel<0><<<gD, 256, 0, stream>>>(mmr, Wmr16, nullptr, rr, nullptr,
                                              (int)N, D, D);

  // 9) out = x2 + sigmoid(rr)*mlp
  final_kernel<<<EW, 256, 0, stream>>>(x2, rr, mlp, out);
}
